// PWCNetAlignment_47863115546773
// MI455X (gfx1250) — compile-verified, matrix operand order checked by reading, not run on hardware
//
#include <hip/hip_runtime.h>
#include <hip/hip_bf16.h>

// ============================================================================
// PWCNet-style multi-frame alignment for MI455X (gfx1250).
//
//  * All 3x3 convolutions + the deformable-conv einsum are implicit GEMMs on
//    v_wmma_f32_16x16x32_f16 (f16 in, f32 accumulate). K-depths 441..2745 ->
//    compute-bound, so the matrix pipe is the only path to the ceiling.
//  * Block tile 64 pixels x 64 out-channels x K32, 4 waves, 4 WMMA
//    accumulators per wave: A tile (the expensive gather) is staged once per
//    64 output channels, and each barrier interval issues 16 WMMAs per block.
//  * All spatial dims are powers of two -> address math uses shifts/masks
//    (no runtime integer division in the hot loops).
//  * Everything else (corr, normalize, resize, pooling, casts) moves < 2 GB
//    total against 23.3 TB/s HBM -> bandwidth-trivial VALU kernels.
//
// WMMA operand layout follows ISA 7.12.2 (16-bit A 16x32: lanes 0-15 hold
// K groups {0..7,16..23}, lanes 16-31 hold {8..15,24..31}); B assumed
// symmetric with N = lane%16. Compile-only environment.
// ============================================================================

typedef __attribute__((ext_vector_type(16))) _Float16 v16h;
typedef __attribute__((ext_vector_type(8)))  _Float16 v8h;
typedef __attribute__((ext_vector_type(8)))  float    v8f;

#define NEG_SLOPE 0.1f
#define BM 64
#define BN 64
#define BK 32
#define LDA 40   // padded LDS stride (halves); 80B keeps 16B alignment, spreads banks
#define LDB 40

__device__ __forceinline__ float lrelu_f(float v) { return v >= 0.f ? v : NEG_SLOPE * v; }

// ---------------------------------------------------------------------------
// Elementwise / data-movement kernels (bandwidth-bound)
// ---------------------------------------------------------------------------

__global__ void k_downsample2(const float* __restrict__ in, float* __restrict__ out,
                              int M, int Ho, int Wo) {
  int idx = blockIdx.x * blockDim.x + threadIdx.x;
  int tot = M * Ho * Wo;
  if (idx >= tot) return;
  int m  = idx / (Ho * Wo);
  int r  = idx - m * (Ho * Wo);
  int oy = r / Wo, ox = r - oy * Wo;
  int Wi = 2 * Wo;
  const float* ib = in + (size_t)m * (4 * Ho * Wo);
  float s = ib[(2 * oy) * Wi + 2 * ox]     + ib[(2 * oy) * Wi + 2 * ox + 1]
          + ib[(2 * oy + 1) * Wi + 2 * ox] + ib[(2 * oy + 1) * Wi + 2 * ox + 1];
  out[idx] = 0.25f * s;
}

__global__ void k_slice_t(const float* __restrict__ x, float* __restrict__ out,
                          int B, int T, int ti, int CHW) {
  int idx = blockIdx.x * blockDim.x + threadIdx.x;
  if (idx >= B * CHW) return;
  int bb = idx / CHW, r = idx - bb * CHW;
  out[idx] = x[((size_t)bb * T + ti) * CHW + r];
}

__global__ void k_copy_center(const float* __restrict__ x, float* __restrict__ out,
                              int B, int T, int CHW) {
  int idx = blockIdx.x * blockDim.x + threadIdx.x;
  if (idx >= B * CHW) return;
  int bb = idx / CHW, r = idx - bb * CHW;
  size_t o = ((size_t)bb * T + 1) * CHW + r;
  out[o] = x[o];
}

__global__ void k_normalize(const float* __restrict__ f, float* __restrict__ fn,
                            int B, int C, int HW) {
  int idx = blockIdx.x * blockDim.x + threadIdx.x;
  if (idx >= B * HW) return;
  int bb = idx / HW, p = idx - bb * HW;
  const float* base = f + (size_t)bb * C * HW + p;
  float ss = 0.f;
  for (int c = 0; c < C; ++c) { float v = base[(size_t)c * HW]; ss += v * v; }
  float inv = 1.f / (sqrtf(ss) + 1e-6f);
  float* ob = fn + (size_t)bb * C * HW + p;
  for (int c = 0; c < C; ++c) ob[(size_t)c * HW] = base[(size_t)c * HW] * inv;
}

// out(b, d, y, x) = lrelu( sum_c f1n[c, y+dy-3, x+dx-3] * f2n[c, y, x] )
__global__ void k_corr(const float* __restrict__ f1n, const float* __restrict__ f2n,
                       float* __restrict__ out, int B, int C, int H, int W) {
  int HW = H * W;
  int idx = blockIdx.x * blockDim.x + threadIdx.x;
  if (idx >= B * 49 * HW) return;
  int bb = idx / (49 * HW);
  int r  = idx - bb * 49 * HW;
  int d  = r / HW;
  int p  = r - d * HW;
  int y = p / W, x = p - y * W;
  int y1 = y + d / 7 - 3;
  int x1 = x + d % 7 - 3;
  float s = 0.f;
  if (y1 >= 0 && y1 < H && x1 >= 0 && x1 < W) {
    const float* a  = f1n + (size_t)bb * C * HW + y1 * W + x1;
    const float* b2 = f2n + (size_t)bb * C * HW + p;
    for (int c = 0; c < C; ++c) s += a[(size_t)c * HW] * b2[(size_t)c * HW];
  }
  out[idx] = lrelu_f(s);
}

__global__ void k_cast_slice(const float* __restrict__ src, int sCt, int sCo,
                             _Float16* __restrict__ dst, int dCt, int dCo,
                             int Cn, int B, int HW) {
  int idx = blockIdx.x * blockDim.x + threadIdx.x;
  if (idx >= B * Cn * HW) return;
  int bb = idx / (Cn * HW);
  int c  = (idx / HW) % Cn;
  int p  = idx % HW;
  dst[((size_t)bb * dCt + dCo + c) * HW + p] =
      (_Float16)src[((size_t)bb * sCt + sCo + c) * HW + p];
}

// half-pixel bilinear x2 upsample (matches jax.image.resize 'bilinear'), with scale
__global__ void k_upsample2(const float* __restrict__ in, float* __restrict__ out,
                            int B, int Cn, int Hi, int Wi, float sc) {
  int Ho = 2 * Hi, Wo = 2 * Wi;
  int idx = blockIdx.x * blockDim.x + threadIdx.x;
  int tot = B * Cn * Ho * Wo;
  if (idx >= tot) return;
  int map = idx / (Ho * Wo);
  int r   = idx - map * (Ho * Wo);
  int oy = r / Wo, ox = r - oy * Wo;
  float fy = (oy + 0.5f) * 0.5f - 0.5f;
  float fx = (ox + 0.5f) * 0.5f - 0.5f;
  float y0f = floorf(fy), x0f = floorf(fx);
  int y0 = (int)y0f, x0 = (int)x0f;
  float wy = fy - y0f, wx = fx - x0f;
  int ya = min(max(y0, 0), Hi - 1), yb = min(max(y0 + 1, 0), Hi - 1);
  int xa = min(max(x0, 0), Wi - 1), xb = min(max(x0 + 1, 0), Wi - 1);
  const float* mb = in + (size_t)map * Hi * Wi;
  float v = mb[ya * Wi + xa] * (1.f - wy) * (1.f - wx)
          + mb[ya * Wi + xb] * (1.f - wy) * wx
          + mb[yb * Wi + xa] * wy * (1.f - wx)
          + mb[yb * Wi + xb] * wy * wx;
  out[idx] = sc * v;
}

// pack (cout, cin, 3, 3) f32 weights -> f16 [K][cout], K = cin*9, k = ci*9 + ky*3 + kx
__global__ void k_pack_w(const float* __restrict__ w, _Float16* __restrict__ bp,
                         int cout, int cin) {
  int K = cin * 9;
  int idx = blockIdx.x * blockDim.x + threadIdx.x;
  if (idx >= K * cout) return;
  int k = idx / cout, n = idx - k * cout;
  int ci = k / 9, r = k - ci * 9;
  bp[idx] = (_Float16)w[((size_t)n * cin + ci) * 9 + r];
}

__global__ void k_add(const float* __restrict__ a, const float* __restrict__ b,
                      float* __restrict__ out, int n) {
  int idx = blockIdx.x * blockDim.x + threadIdx.x;
  if (idx < n) out[idx] = a[idx] + b[idx];
}

// ---------------------------------------------------------------------------
// WMMA helpers
// ---------------------------------------------------------------------------

__device__ __forceinline__ v16h load_frag(const _Float16* row, int klo) {
  v8h lo = *(const v8h*)(row + klo);
  v8h hi = *(const v8h*)(row + klo + 16);
  v16h a;
#pragma unroll
  for (int h = 0; h < 8; ++h) { a[h] = lo[h]; a[8 + h] = hi[h]; }
  return a;
}

// ---------------------------------------------------------------------------
// Implicit-GEMM 3x3 conv (pad=1) via v_wmma_f32_16x16x32_f16.
// Block: 128 threads (4 waves). Tile: 64 pixels x 64 out-channels, K-step 32.
// Wave w computes M rows [16w, 16w+16) x all 64 N cols (4 accumulators).
// H == W == 2^sh (power of two): address math is shift/mask only.
// Epilogue: bias + optional leaky-ReLU, write into channel slice of `out`.
// ---------------------------------------------------------------------------
__global__ __launch_bounds__(128)
void k_conv_wmma(const _Float16* __restrict__ act, const _Float16* __restrict__ Bp,
                 const float* __restrict__ bias, float* __restrict__ out,
                 int Cin, int Cout, int sh,
                 int outCtot, int outCoff, int doAct) {
  __shared__ _Float16 As[BM * LDA];
  __shared__ _Float16 Bs[BN * LDB];
  const int tid  = threadIdx.x;
  const int wave = tid >> 5;
  const int lane = tid & 31;
  const int bM   = blockIdx.x * BM;
  const int bN   = blockIdx.y * BN;
  const int bb   = blockIdx.z;
  const int W    = 1 << sh;
  const int H    = W;
  const int HW   = H * W;
  const int wm   = W - 1;
  const int K    = Cin * 9;
  const _Float16* actB = act + (size_t)bb * Cin * HW;

  v8f acc[4];
#pragma unroll
  for (int j = 0; j < 4; ++j) acc[j] = (v8f){0.f, 0.f, 0.f, 0.f, 0.f, 0.f, 0.f, 0.f};

  for (int k0 = 0; k0 < K; k0 += BK) {
    // --- stage A tile (im2col gather, zero-padded) : 64x32 ---
#pragma unroll
    for (int i = 0; i < 16; ++i) {
      int idx = tid + i * 128;
      int m = idx >> 5, kk = idx & 31;
      int kg = k0 + kk;
      _Float16 v = (_Float16)0.f;
      if (kg < K) {
        int ci = kg / 9, r = kg - ci * 9;   // constant divisor -> mul/shift
        int p = bM + m;
        int y = p >> sh, x = p & wm;
        int yy = y + r / 3 - 1, xx = x + r % 3 - 1;
        if (yy >= 0 && yy < H && xx >= 0 && xx < W)
          v = actB[(size_t)ci * HW + (yy << sh) + xx];
      }
      As[m * LDA + kk] = v;
    }
    // --- stage B tile transposed to [n][k] : 64x32 ---
#pragma unroll
    for (int i = 0; i < 16; ++i) {
      int idx = tid + i * 128;
      int kk = (idx >> 6), n = idx & 63;      // 64 n x 32 k, two k per thread-pass
      kk = kk + ((i & 1) ? 0 : 0);            // layout: idx = n + kk*64
      int kg = k0 + kk, gn = bN + n;
      _Float16 v = (_Float16)0.f;
      if (kg < K && gn < Cout) v = Bp[(size_t)kg * Cout + gn];
      Bs[n * LDB + kk] = v;
    }
    if (tid == 0 && k0 + BK < K)
      __builtin_prefetch(Bp + (size_t)(k0 + BK) * Cout + bN, 0, 0);  // global_prefetch_b8
    __syncthreads();

    const int klo  = (lane < 16) ? 0 : 8;
    const int mrow = wave * 16 + (lane & 15);
    const int ncol = lane & 15;
    v16h a = load_frag(&As[mrow * LDA], klo);
#pragma unroll
    for (int j = 0; j < 4; ++j) {
      v16h bf = load_frag(&Bs[(j * 16 + ncol) * LDB], klo);
      acc[j] = __builtin_amdgcn_wmma_f32_16x16x32_f16(false, a, false, bf, (short)0, acc[j], false, false);
    }
    __syncthreads();
  }

  const int mbase = bM + wave * 16 + ((lane < 16) ? 0 : 8);
  const int ncol  = lane & 15;
#pragma unroll
  for (int j = 0; j < 4; ++j) {
    int n = bN + j * 16 + ncol;
    if (n >= Cout) continue;
    float bv = bias[n];
    float* ob = out + ((size_t)bb * outCtot + outCoff + n) * HW;
#pragma unroll
    for (int r = 0; r < 8; ++r) {
      int m = mbase + r;
      float v = acc[j][r] + bv;
      if (doAct) v = lrelu_f(v);
      ob[((m >> sh) << sh) + (m & wm)] = v;
    }
  }
}

// ---------------------------------------------------------------------------
// Modulated deformable conv (DG=8, cper=8, kk=9) as WMMA GEMM with K=576.
// A(p, kg) with kg = ci*9 + r, ci = g*8 + c: bilinear sample of x at
// (y-1+ky+off_y, x-1+kx+off_x), per-corner zero padding, times sigmoid(mask).
// Weight pack identical to the normal conv pack (wr[o,g,c,k] = w[o,g*8+c,ky,kx]).
// Cout = 64 -> single N tile, 4 accumulators per wave.
// ---------------------------------------------------------------------------
__global__ __launch_bounds__(128)
void k_dconv_wmma(const float* __restrict__ xin, const float* __restrict__ offr,
                  const _Float16* __restrict__ Bp, const float* __restrict__ bias,
                  float* __restrict__ out, int sh, int outCtot, int outCoff) {
  const int K = 576, Cout = 64;
  __shared__ _Float16 As[BM * LDA];
  __shared__ _Float16 Bs[BN * LDB];
  const int tid  = threadIdx.x;
  const int wave = tid >> 5;
  const int lane = tid & 31;
  const int bM   = blockIdx.x * BM;
  const int bb   = blockIdx.z;
  const int W    = 1 << sh;
  const int H    = W;
  const int HW   = H * W;
  const int wm   = W - 1;
  const float* xb = xin  + (size_t)bb * 64  * HW;
  const float* ob = offr + (size_t)bb * 216 * HW;

  v8f acc[4];
#pragma unroll
  for (int j = 0; j < 4; ++j) acc[j] = (v8f){0.f, 0.f, 0.f, 0.f, 0.f, 0.f, 0.f, 0.f};

  for (int k0 = 0; k0 < K; k0 += BK) {
#pragma unroll
    for (int i = 0; i < 16; ++i) {
      int idx = tid + i * 128;
      int m = idx >> 5, kk = idx & 31;
      int kg = k0 + kk;
      int ci = kg / 9, r = kg - ci * 9;
      int g  = ci >> 3;
      int p  = bM + m;
      int y = p >> sh, x = p & wm;
      int oc = (g * 9 + r) * 2;
      float oy = ob[(size_t)oc * HW + p];
      float ox = ob[(size_t)(oc + 1) * HW + p];
      float mv = ob[(size_t)(144 + g * 9 + r) * HW + p];
      mv = 1.f / (1.f + expf(-mv));
      float py = (float)y - 1.f + (float)(r / 3) + oy;
      float px = (float)x - 1.f + (float)(r % 3) + ox;
      float y0f = floorf(py), x0f = floorf(px);
      int y0 = (int)y0f, x0 = (int)x0f;
      float wy = py - y0f, wx = px - x0f;
      const float* ch = xb + (size_t)ci * HW;
      float v00 = (y0 >= 0 && y0 < H && x0 >= 0 && x0 < W)         ? ch[(y0 << sh) + x0]           : 0.f;
      float v01 = (y0 >= 0 && y0 < H && x0 + 1 >= 0 && x0 + 1 < W) ? ch[(y0 << sh) + x0 + 1]       : 0.f;
      float v10 = (y0 + 1 >= 0 && y0 + 1 < H && x0 >= 0 && x0 < W) ? ch[((y0 + 1) << sh) + x0]     : 0.f;
      float v11 = (y0 + 1 >= 0 && y0 + 1 < H && x0 + 1 >= 0 && x0 + 1 < W) ? ch[((y0 + 1) << sh) + x0 + 1] : 0.f;
      float s = v00 * (1.f - wy) * (1.f - wx) + v01 * (1.f - wy) * wx
              + v10 * wy * (1.f - wx)         + v11 * wy * wx;
      As[m * LDA + kk] = (_Float16)(s * mv);
    }
#pragma unroll
    for (int i = 0; i < 16; ++i) {
      int idx = tid + i * 128;
      int kk = idx >> 6, n = idx & 63;
      Bs[n * LDB + kk] = Bp[(size_t)(k0 + kk) * Cout + n];
    }
    __syncthreads();

    const int klo  = (lane < 16) ? 0 : 8;
    const int mrow = wave * 16 + (lane & 15);
    const int ncol = lane & 15;
    v16h a = load_frag(&As[mrow * LDA], klo);
#pragma unroll
    for (int j = 0; j < 4; ++j) {
      v16h bf = load_frag(&Bs[(j * 16 + ncol) * LDB], klo);
      acc[j] = __builtin_amdgcn_wmma_f32_16x16x32_f16(false, a, false, bf, (short)0, acc[j], false, false);
    }
    __syncthreads();
  }

  const int mbase = bM + wave * 16 + ((lane < 16) ? 0 : 8);
  const int ncol  = lane & 15;
#pragma unroll
  for (int j = 0; j < 4; ++j) {
    int n = j * 16 + ncol;
    float bv = bias[n];
    float* op = out + ((size_t)bb * outCtot + outCoff + n) * HW;
#pragma unroll
    for (int r = 0; r < 8; ++r) {
      int m = mbase + r;
      op[m] = acc[j][r] + bv;
    }
  }
}

// ---------------------------------------------------------------------------
// Host orchestration
// ---------------------------------------------------------------------------

#define LAUNCH1(kname, n, ...) \
  kname<<<dim3((unsigned)(((n) + 255) / 256)), dim3(256), 0, stream>>>(__VA_ARGS__)

enum { P_OA3 = 0, P_OB3, P_OC3, P_OA2, P_OB2, P_OC2, P_OA1, P_OB1, P_OC1,
       P_DCN2_OFF, P_DCN2, P_DCN1_OFF, P_DCN1, P_DCNF_OFF, P_DCNF,
       P_REF1, P_REF2, P_REF3, P_COUNT };

extern "C" void kernel_launch(void* const* d_in, const int* in_sizes, int n_in,
                              void* d_out, int out_size, void* d_ws, size_t ws_size,
                              hipStream_t stream) {
  (void)in_sizes; (void)out_size;
  if (n_in < 1 + 2 * P_COUNT) return;

  constexpr int Bn = 2, T = 3, Cc = 64;
  constexpr int H1 = 128, W1 = 128, HW1 = H1 * W1, SH1 = 7;
  constexpr int H2 = 64,  W2 = 64,  HW2 = H2 * W2, SH2 = 6;
  constexpr int H3 = 32,  W3 = 32,  HW3 = H3 * W3, SH3 = 5;

  const float* X = (const float*)d_in[0];
  const float* Wg[P_COUNT]; const float* Bi[P_COUNT];
  for (int i = 0; i < P_COUNT; ++i) {
    Wg[i] = (const float*)d_in[1 + 2 * i];
    Bi[i] = (const float*)d_in[2 + 2 * i];
  }
  static const int cinA[P_COUNT]  = {49, 64, 128, 305, 64, 128, 305, 64, 128,
                                     64, 64, 64, 64, 64, 64, 128, 64, 64};
  static const int coutA[P_COUNT] = {64, 64, 64, 64, 64, 64, 64, 64, 64,
                                     216, 64, 216, 64, 216, 64, 64, 64, 64};

  // ---- workspace bump allocator (host arithmetic only) ----
  char* ws = (char*)d_ws;
  size_t off = 0;
  auto allocB = [&](size_t bytes) -> void* {
    off = (off + 255) & ~(size_t)255;
    void* p = ws + off;
    off += bytes;
    return p;
  };
  auto aF = [&](size_t n) { return (float*)allocB(n * sizeof(float)); };
  auto aH = [&](size_t n) { return (_Float16*)allocB(n * sizeof(_Float16)); };

  float* x2   = aF((size_t)Bn * T * Cc * HW2);
  float* x3   = aF((size_t)Bn * T * Cc * HW3);
  float* c1   = aF((size_t)Bn * Cc * HW1);
  float* c2   = aF((size_t)Bn * Cc * HW2);
  float* c3   = aF((size_t)Bn * Cc * HW3);
  float* n1   = aF((size_t)Bn * Cc * HW1);
  float* n2   = aF((size_t)Bn * Cc * HW2);
  float* n3   = aF((size_t)Bn * Cc * HW3);
  float* c1n  = aF((size_t)Bn * Cc * HW1);
  float* c2n  = aF((size_t)Bn * Cc * HW2);
  float* c3n  = aF((size_t)Bn * Cc * HW3);
  float* nn3  = aF((size_t)Bn * Cc * HW3);
  float* fa2n = aF((size_t)Bn * Cc * HW2);
  float* fa1n = aF((size_t)Bn * Cc * HW1);
  float* cost3 = aF((size_t)Bn * 49 * HW3);
  float* cost2 = aF((size_t)Bn * 49 * HW2);
  float* cost1 = aF((size_t)Bn * 49 * HW1);
  float* cf3   = aF((size_t)Bn * 128 * HW3);
  float* off3  = aF((size_t)Bn * 64 * HW3);
  float* u_cf3 = aF((size_t)Bn * 128 * HW2);
  float* u_off3 = aF((size_t)Bn * 64 * HW2);
  float* off2raw = aF((size_t)Bn * 216 * HW2);
  float* fa2   = aF((size_t)Bn * 64 * HW2);
  float* cf2   = aF((size_t)Bn * 128 * HW2);
  float* off2  = aF((size_t)Bn * 64 * HW2);
  float* u_cf2 = aF((size_t)Bn * 128 * HW1);
  float* u_off2 = aF((size_t)Bn * 64 * HW1);
  float* off1raw = aF((size_t)Bn * 216 * HW1);
  float* fa1   = aF((size_t)Bn * 64 * HW1);
  float* cf1   = aF((size_t)Bn * 128 * HW1);
  float* off1  = aF((size_t)Bn * 64 * HW1);
  float* r1    = aF((size_t)Bn * 64 * HW1);
  float* r2    = aF((size_t)Bn * 64 * HW1);
  float* r3    = aF((size_t)Bn * 64 * HW1);
  float* offsum = aF((size_t)Bn * 64 * HW1);
  float* offFraw = aF((size_t)Bn * 216 * HW1);

  _Float16* cost3h = aH((size_t)Bn * 49 * HW3);
  _Float16* cf3ah  = aH((size_t)Bn * 64 * HW3);
  _Float16* cf3h   = aH((size_t)Bn * 128 * HW3);
  _Float16* uoff3h = aH((size_t)Bn * 64 * HW2);
  _Float16* cat2h  = aH((size_t)Bn * 305 * HW2);
  _Float16* cf2ah  = aH((size_t)Bn * 64 * HW2);
  _Float16* cf2h   = aH((size_t)Bn * 128 * HW2);
  _Float16* uoff2h = aH((size_t)Bn * 64 * HW1);
  _Float16* cat1h  = aH((size_t)Bn * 305 * HW1);
  _Float16* cf1ah  = aH((size_t)Bn * 64 * HW1);
  _Float16* cf1h   = aH((size_t)Bn * 128 * HW1);
  _Float16* r1h    = aH((size_t)Bn * 64 * HW1);
  _Float16* r2h    = aH((size_t)Bn * 64 * HW1);
  _Float16* offsumh = aH((size_t)Bn * 64 * HW1);

  _Float16* wpack[P_COUNT];
  for (int i = 0; i < P_COUNT; ++i)
    wpack[i] = aH((size_t)cinA[i] * 9 * coutA[i]);

  if (off > ws_size) return;  // workspace too small: deterministic no-op

  float* outF = (float*)d_out;

  // ---- helper lambdas ----
  auto conv = [&](const _Float16* act, int Cin, int pidx, float* out,
                  int HW, int sh, int Ctot, int Coff, int doAct) {
    int Cout = coutA[pidx];
    dim3 g((unsigned)(HW / BM), (unsigned)((Cout + BN - 1) / BN), (unsigned)Bn);
    k_conv_wmma<<<g, 128, 0, stream>>>(act, wpack[pidx], Bi[pidx], out,
                                       Cin, Cout, sh, Ctot, Coff, doAct);
  };
  auto dconv = [&](const float* xin, const float* offr, int pidx, float* out,
                   int HW, int sh, int Ctot, int Coff) {
    dim3 g((unsigned)(HW / BM), 1u, (unsigned)Bn);
    k_dconv_wmma<<<g, 128, 0, stream>>>(xin, offr, wpack[pidx], Bi[pidx], out,
                                        sh, Ctot, Coff);
  };
  auto cast = [&](const float* src, int sCt, int sCo, _Float16* dst, int dCt, int dCo,
                  int Cn, int HW) {
    LAUNCH1(k_cast_slice, Bn * Cn * HW, src, sCt, sCo, dst, dCt, dCo, Cn, Bn, HW);
  };
  auto ups = [&](const float* in, float* out, int Cn, int Hi, int Wi, float sc) {
    LAUNCH1(k_upsample2, Bn * Cn * 4 * Hi * Wi, in, out, Bn, Cn, Hi, Wi, sc);
  };
  auto nrm = [&](const float* f, float* fn, int HW) {
    LAUNCH1(k_normalize, Bn * HW, f, fn, Bn, Cc, HW);
  };
  auto corrk = [&](const float* a, const float* b2, float* o, int H, int W) {
    LAUNCH1(k_corr, Bn * 49 * H * W, a, b2, o, Bn, Cc, H, W);
  };

  // ---- one-time prep ----
  for (int i = 0; i < P_COUNT; ++i) {
    int tot = cinA[i] * 9 * coutA[i];
    LAUNCH1(k_pack_w, tot, Wg[i], wpack[i], coutA[i], cinA[i]);
  }
  LAUNCH1(k_downsample2, Bn * T * Cc * HW2, X,  x2, Bn * T * Cc, H2, W2);
  LAUNCH1(k_downsample2, Bn * T * Cc * HW3, x2, x3, Bn * T * Cc, H3, W3);
  LAUNCH1(k_slice_t, Bn * Cc * HW1, X,  c1, Bn, T, 1, Cc * HW1);
  LAUNCH1(k_slice_t, Bn * Cc * HW2, x2, c2, Bn, T, 1, Cc * HW2);
  LAUNCH1(k_slice_t, Bn * Cc * HW3, x3, c3, Bn, T, 1, Cc * HW3);
  LAUNCH1(k_copy_center, Bn * Cc * HW1, X, outF, Bn, T, Cc * HW1);
  nrm(c1, c1n, HW1);
  nrm(c2, c2n, HW2);
  nrm(c3, c3n, HW3);

  // ---- align_one for the two non-center frames ----
  for (int pass = 0; pass < 2; ++pass) {
    int ti = (pass == 0) ? 0 : 2;
    LAUNCH1(k_slice_t, Bn * Cc * HW3, x3, n3, Bn, T, ti, Cc * HW3);
    LAUNCH1(k_slice_t, Bn * Cc * HW2, x2, n2, Bn, T, ti, Cc * HW2);
    LAUNCH1(k_slice_t, Bn * Cc * HW1, X,  n1, Bn, T, ti, Cc * HW1);

    // level 3
    nrm(n3, nn3, HW3);
    corrk(c3n, nn3, cost3, H3, W3);
    cast(cost3, 49, 0, cost3h, 49, 0, 49, HW3);
    conv(cost3h, 49, P_OA3, cf3, HW3, SH3, 128, 64, 1);
    cast(cf3, 128, 64, cf3ah, 64, 0, 64, HW3);
    conv(cf3ah, 64, P_OB3, cf3, HW3, SH3, 128, 0, 1);
    cast(cf3, 128, 0, cf3h, 128, 0, 128, HW3);
    conv(cf3h, 128, P_OC3, off3, HW3, SH3, 64, 0, 0);
    ups(cf3, u_cf3, 128, H3, W3, 1.f);
    ups(off3, u_off3, 64, H3, W3, 2.f);

    // level 2: mdcn + correlation + conv stack
    cast(u_off3, 64, 0, uoff3h, 64, 0, 64, HW2);
    conv(uoff3h, 64, P_DCN2_OFF, off2raw, HW2, SH2, 216, 0, 0);
    dconv(n2, off2raw, P_DCN2, fa2, HW2, SH2, 64, 0);
    nrm(fa2, fa2n, HW2);
    corrk(c2n, fa2n, cost2, H2, W2);
    cast(cost2, 49, 0, cat2h, 305, 0, 49, HW2);
    cast(c2, 64, 0, cat2h, 305, 49, 64, HW2);
    cast(u_off3, 64, 0, cat2h, 305, 113, 64, HW2);
    cast(u_cf3, 128, 0, cat2h, 305, 177, 128, HW2);
    conv(cat2h, 305, P_OA2, cf2, HW2, SH2, 128, 64, 1);
    cast(cf2, 128, 64, cf2ah, 64, 0, 64, HW2);
    conv(cf2ah, 64, P_OB2, cf2, HW2, SH2, 128, 0, 1);
    cast(cf2, 128, 0, cf2h, 128, 0, 128, HW2);
    conv(cf2h, 128, P_OC2, off2, HW2, SH2, 64, 0, 0);
    ups(cf2, u_cf2, 128, H2, W2, 1.f);
    ups(off2, u_off2, 64, H2, W2, 2.f);

    // level 1: mdcn + correlation + conv stack + refinement
    cast(u_off2, 64, 0, uoff2h, 64, 0, 64, HW1);
    conv(uoff2h, 64, P_DCN1_OFF, off1raw, HW1, SH1, 216, 0, 0);
    dconv(n1, off1raw, P_DCN1, fa1, HW1, SH1, 64, 0);
    nrm(fa1, fa1n, HW1);
    corrk(c1n, fa1n, cost1, H1, W1);
    cast(cost1, 49, 0, cat1h, 305, 0, 49, HW1);
    cast(c1, 64, 0, cat1h, 305, 49, 64, HW1);
    cast(u_off2, 64, 0, cat1h, 305, 113, 64, HW1);
    cast(u_cf2, 128, 0, cat1h, 305, 177, 128, HW1);
    conv(cat1h, 305, P_OA1, cf1, HW1, SH1, 128, 64, 1);
    cast(cf1, 128, 64, cf1ah, 64, 0, 64, HW1);
    conv(cf1ah, 64, P_OB1, cf1, HW1, SH1, 128, 0, 1);
    cast(cf1, 128, 0, cf1h, 128, 0, 128, HW1);
    conv(cf1h, 128, P_OC1, off1, HW1, SH1, 64, 0, 0);
    conv(cf1h, 128, P_REF1, r1, HW1, SH1, 64, 0, 1);
    cast(r1, 64, 0, r1h, 64, 0, 64, HW1);
    conv(r1h, 64, P_REF2, r2, HW1, SH1, 64, 0, 1);
    cast(r2, 64, 0, r2h, 64, 0, 64, HW1);
    conv(r2h, 64, P_REF3, r3, HW1, SH1, 64, 0, 0);
    LAUNCH1(k_add, Bn * 64 * HW1, off1, r3, offsum, Bn * 64 * HW1);

    // final mdcn writes straight into d_out, viewed as (b, T*64, H, W)
    cast(offsum, 64, 0, offsumh, 64, 0, 64, HW1);
    conv(offsumh, 64, P_DCNF_OFF, offFraw, HW1, SH1, 216, 0, 0);
    dconv(n1, offFraw, P_DCNF, outF, HW1, SH1, T * Cc, ti * Cc);
  }
}
